// TextRNN_83245056131326
// MI455X (gfx1250) — matrix operand order repaired and checked by reading, not compiled, not run on hardware
//
#include <hip/hip_runtime.h>
#include <hip/hip_bf16.h>

#define Bb 64
#define Tt 256
#define Ee 512
#define Hh 512
#define OUTW (2*Hh + Ee)   // 1536

typedef unsigned int u32;
typedef __bf16 bf16;
typedef __attribute__((ext_vector_type(16))) __bf16 v16bf;
typedef __attribute__((ext_vector_type(8)))  float  v8f;

union Frag16 { v16bf v; uint4 q[2]; };

static __device__ inline u32 pk2(float a, float b) {
    union { __bf16 h[2]; u32 u; } x;
    x.h[0] = (__bf16)a; x.h[1] = (__bf16)b;
    return x.u;
}

// ---------------------------------------------------------------------------
// Kernel 1: pack a fp32 weight matrix W[E][H] (row-major, K x N) into wave32
// WMMA bf16 B-fragment tiles. Tile (kt, nt) covers K = kt*32..+31, N = nt*16..+15.
// Lane l: column n = nt*16 + (l&15); K-base = kt*32 + (l>>4)*8, elems 8..15 at +16; 16 bf16 values
// packed pairwise -> 32 bytes per lane, 1024 B per tile, tiles stored kt*32+nt.
// grid = (512 tiles, 4 weights), block = 32 (one wave).
// ---------------------------------------------------------------------------
__global__ void pack_weights_kernel(const float* w0, const float* w1,
                                    const float* w2, const float* w3,
                                    bf16* Wpk) {
    const float* Ws[4] = {w0, w1, w2, w3};
    const float* W = Ws[blockIdx.y];
    int tIdx = blockIdx.x;               // kt*32 + nt
    int kt = tIdx >> 5;
    int nt = tIdx & 31;
    int lane = threadIdx.x & 31;
    int n  = nt * 16 + (lane & 15);
    int kb = kt * 32 + (lane >> 4) * 8;

    u32 packed[8];
#pragma unroll
    for (int j = 0; j < 8; ++j) {
        float a = W[(size_t)(kb + 2*j + (j >> 2) * 8    ) * Hh + n];
        float b = W[(size_t)(kb + 2*j + (j >> 2) * 8 + 1) * Hh + n];
        packed[j] = pk2(a, b);
    }
    uint4* dq = (uint4*)(Wpk + (size_t)blockIdx.y * Ee * Hh + (size_t)tIdx * 512 + lane * 16);
    dq[0] = make_uint4(packed[0], packed[1], packed[2], packed[3]);
    dq[1] = make_uint4(packed[4], packed[5], packed[6], packed[7]);
}

// ---------------------------------------------------------------------------
// Kernel 2: convert left_state / right_state to bf16. stateBF[2][B][E].
// ---------------------------------------------------------------------------
__global__ void states_kernel(const float* ls, const float* rs, bf16* stateBF) {
    int idx = (blockIdx.x * blockDim.x + threadIdx.x) * 4;   // 0..65535
    const int half = Bb * Ee;                                // 32768
    const float* src = (idx < half) ? (ls + idx) : (rs + (idx - half));
    float4 f = *(const float4*)src;
    uint2 p;
    p.x = pk2(f.x, f.y);
    p.y = pk2(f.z, f.w);
    *(uint2*)(stateBF + idx) = p;
}

// ---------------------------------------------------------------------------
// Kernel 3: embedding gather -> output middle slice (fp32) + embBF (bf16).
// grid = (T, B), block = 128, float4 per thread.
// ---------------------------------------------------------------------------
__global__ void gather_kernel(const int* tokens, const float* emb,
                              float* out, bf16* embBF) {
    int t = blockIdx.x, b = blockIdx.y;
    int tok = tokens[b * Tt + t];
    int i = threadIdx.x;
    float4 e = ((const float4*)(emb + (size_t)tok * Ee))[i];

    float* orow = out + ((size_t)b * Tt + t) * OUTW + Hh;
    ((float4*)orow)[i] = e;

    uint2 p;
    p.x = pk2(e.x, e.y);
    p.y = pk2(e.z, e.w);
    *(uint2*)(embBF + ((size_t)b * Tt + t) * Ee + i * 4) = p;
}

// ---------------------------------------------------------------------------
// Kernel 4: pre-GEMM  P[s][m][n] = X_s[m] @ W1_s   (M=B*T, K=E, N=H)
//   X_l[b,t] = (t==0) ? left_state[b]  : emb[b,t-1]
//   X_r[b,t] = (t==0) ? right_state[b] : emb[b,T-t]
// grid = (1024, 2), block = 256 (8 waves). Wave w: 16 x 64 slice.
// Double-buffered A and B fragments so WMMAs overlap the next k-step's loads.
// ---------------------------------------------------------------------------
__global__ void __launch_bounds__(256) pregemm_kernel(
        const bf16* embBF, const bf16* stateBF, const bf16* Wpk, float* P) {
    int mtile = blockIdx.x;
    int s = blockIdx.y;
    int lane = threadIdx.x & 31;
    int wave = threadIdx.x >> 5;

    int m0 = mtile * 16;
    int b  = m0 / Tt;
    int t0 = m0 % Tt;            // tiles never cross batch rows (T % 16 == 0)
    int row = lane & 15;
    int sel = lane >> 4;
    int t = t0 + row;

    const bf16* arow;
    if (t == 0)      arow = stateBF + ((size_t)s * Bb + b) * Ee;
    else if (s == 0) arow = embBF + ((size_t)b * Tt + (t - 1)) * Ee;
    else             arow = embBF + ((size_t)b * Tt + (Tt - t)) * Ee;

    const bf16* wbase = Wpk + (size_t)(s * 2) * Ee * Hh;   // W_l1 (idx0) / W_r1 (idx2)
    int ntg0 = wave * 4;

    Frag16 a[2];
    Frag16 bq[2][4];
    // prologue: k-step 0
    a[0].q[0] = *(const uint4*)(arow + 0 + sel * 8);
    a[0].q[1] = *(const uint4*)(arow + 16 + sel * 8);
#pragma unroll
    for (int nt = 0; nt < 4; ++nt) {
        const uint4* bp = (const uint4*)(wbase + (size_t)(ntg0 + nt) * 512 + lane * 16);
        bq[0][nt].q[0] = bp[0];
        bq[0][nt].q[1] = bp[1];
    }

    v8f acc[4] = {};
#pragma unroll
    for (int kt = 0; kt < Ee / 32; ++kt) {
        int cb = kt & 1, nb = cb ^ 1;
        if (kt < Ee / 32 - 1) {
            int k1 = (kt + 1) * 32;
            a[nb].q[0] = *(const uint4*)(arow + k1 + sel * 8);
            a[nb].q[1] = *(const uint4*)(arow + k1 + 16 + sel * 8);
#pragma unroll
            for (int nt = 0; nt < 4; ++nt) {
                const uint4* bp = (const uint4*)(wbase +
                    (size_t)((kt + 1) * 32 + ntg0 + nt) * 512 + lane * 16);
                bq[nb][nt].q[0] = bp[0];
                bq[nb][nt].q[1] = bp[1];
            }
        }
#pragma unroll
        for (int nt = 0; nt < 4; ++nt) {
            acc[nt] = __builtin_amdgcn_wmma_f32_16x16x32_bf16(
                false, a[cb].v, false, bq[cb][nt].v, (short)0, acc[nt], false, false);
        }
    }

    float* pbase = P + (size_t)s * Bb * Tt * Hh;
#pragma unroll
    for (int nt = 0; nt < 4; ++nt) {
        int n = (ntg0 + nt) * 16 + (lane & 15);
#pragma unroll
        for (int v = 0; v < 8; ++v) {
            int M = v + 8 * sel;
            pbase[(size_t)(m0 + M) * Hh + n] = acc[nt][v];
        }
    }
}

// ---------------------------------------------------------------------------
// Kernel 5: recurrent scan.  ctx_t = tanh(P[:, t, :] + ctx_{t-1} @ W2)
// grid = (4 batch groups, 2 directions), block = 256 (8 waves).
// Latency-oriented schedule per step:
//   1) issue all 32 P loads (overlap the whole WMMA chain)
//   2) load all 16 A-fragments of ctx from LDS (one big ds clause)
//   3) 16-step WMMA chain with double-buffered B tiles from L2
//   4) tanh + store out + store next ctx to LDS, one barrier
// ---------------------------------------------------------------------------
__global__ void __launch_bounds__(256) scan_kernel(
        const float* P, const bf16* Wpk, float* out) {
    __shared__ __align__(16) bf16 ctx[2][16][Hh];   // 32 KB

    int g = blockIdx.x;
    int s = blockIdx.y;
    int lane = threadIdx.x & 31;
    int wave = threadIdx.x >> 5;
    int row = lane & 15;
    int sel = lane >> 4;
    int ntg0 = wave * 4;
    int rowbase = g * 16;
    int colOff = (s == 0) ? 0 : (Hh + Ee);

    {
        u32* z = (u32*)&ctx[0][0][0];
        for (int i = threadIdx.x; i < 16 * Hh / 2; i += blockDim.x) z[i] = 0u;
    }
    __syncthreads();

    const bf16*  wbase = Wpk + (size_t)(s * 2 + 1) * Ee * Hh;  // W_l2 (idx1) / W_r2 (idx3)
    const float* pbase = P + (size_t)s * Bb * Tt * Hh;

    int cur = 0;
    for (int t = 0; t < Tt; ++t) {
        // 1) P loads first: independent of the matmul, hidden under the chain
        float pv[4][8];
#pragma unroll
        for (int nt = 0; nt < 4; ++nt) {
            int n = (ntg0 + nt) * 16 + (lane & 15);
#pragma unroll
            for (int v = 0; v < 8; ++v) {
                int rb = rowbase + v + 8 * sel;
                pv[nt][v] = pbase[((size_t)rb * Tt + t) * Hh + n];
            }
        }

        // 2) all A-fragments of ctx_{t-1}
        Frag16 a[16];
#pragma unroll
        for (int kt = 0; kt < 16; ++kt) {
            a[kt].q[0] = *(const uint4*)&ctx[cur][row][kt * 32 + sel * 8];
            a[kt].q[1] = *(const uint4*)&ctx[cur][row][kt * 32 + 16 + sel * 8];
        }

        // 3) WMMA chain with double-buffered B tiles
        v8f acc[4] = {};
        Frag16 bq[2][4];
#pragma unroll
        for (int nt = 0; nt < 4; ++nt) {
            const uint4* bp = (const uint4*)(wbase + (size_t)(ntg0 + nt) * 512 + lane * 16);
            bq[0][nt].q[0] = bp[0];
            bq[0][nt].q[1] = bp[1];
        }
#pragma unroll
        for (int kt = 0; kt < 16; ++kt) {
            int cb = kt & 1, nb = cb ^ 1;
            if (kt < 15) {
#pragma unroll
                for (int nt = 0; nt < 4; ++nt) {
                    const uint4* bp = (const uint4*)(wbase +
                        (size_t)((kt + 1) * 32 + ntg0 + nt) * 512 + lane * 16);
                    bq[nb][nt].q[0] = bp[0];
                    bq[nb][nt].q[1] = bp[1];
                }
            }
#pragma unroll
            for (int nt = 0; nt < 4; ++nt) {
                acc[nt] = __builtin_amdgcn_wmma_f32_16x16x32_bf16(
                    false, a[kt].v, false, bq[cb][nt].v, (short)0, acc[nt], false, false);
            }
        }

        // 4) epilogue
        int nxt = cur ^ 1;
#pragma unroll
        for (int nt = 0; nt < 4; ++nt) {
            int n = (ntg0 + nt) * 16 + (lane & 15);
#pragma unroll
            for (int v = 0; v < 8; ++v) {
                int M = v + 8 * sel;
                int rb = rowbase + M;
                float val = tanhf(acc[nt][v] + pv[nt][v]);
                out[((size_t)rb * Tt + t) * OUTW + colOff + n] = val;
                ctx[nxt][M][n] = (bf16)val;
            }
        }
        __syncthreads();
        cur = nxt;
    }
}

// ---------------------------------------------------------------------------
// Launch
// ---------------------------------------------------------------------------
extern "C" void kernel_launch(void* const* d_in, const int* in_sizes, int n_in,
                              void* d_out, int out_size, void* d_ws, size_t ws_size,
                              hipStream_t stream) {
    const int*   tokens    = (const int*)  d_in[0];
    const float* embedding = (const float*)d_in[1];
    const float* W_l1      = (const float*)d_in[2];
    const float* W_l2      = (const float*)d_in[3];
    const float* W_r1      = (const float*)d_in[4];
    const float* W_r2      = (const float*)d_in[5];
    const float* lstate    = (const float*)d_in[6];
    const float* rstate    = (const float*)d_in[7];
    float* out = (float*)d_out;

    char* ws = (char*)d_ws;
    bf16* embBF   = (bf16*)ws;                                  // B*T*E bf16  = 16 MB
    bf16* stateBF = embBF + (size_t)Bb * Tt * Ee;               // 2*B*E bf16  = 128 KB
    bf16* Wpk     = stateBF + (size_t)2 * Bb * Ee;              // 4*E*H bf16  = 2 MB
    float* Pbuf   = (float*)(Wpk + (size_t)4 * Ee * Hh);        // 2*B*T*H f32 = 64 MB

    pack_weights_kernel<<<dim3(512, 4), 32, 0, stream>>>(W_l1, W_l2, W_r1, W_r2, Wpk);
    states_kernel<<<64, 256, 0, stream>>>(lstate, rstate, stateBF);
    gather_kernel<<<dim3(Tt, Bb), 128, 0, stream>>>(tokens, embedding, out, embBF);
    pregemm_kernel<<<dim3((Bb * Tt) / 16, 2), 256, 0, stream>>>(embBF, stateBF, Wpk, Pbuf);
    scan_kernel<<<dim3(Bb / 16, 2), 256, 0, stream>>>(Pbuf, Wpk, out);
}
